// T5SelfAttention_33483565040071
// MI455X (gfx1250) — matrix operand order repaired and checked by reading, not compiled, not run on hardware
//
#include <hip/hip_runtime.h>
#include <hip/hip_bf16.h>
#include <math.h>

// ---------------------------------------------------------------------------
// CDNA5 (gfx1250) WMMA types & helpers
// ---------------------------------------------------------------------------
typedef __attribute__((ext_vector_type(16))) _Float16 v16h;
typedef __attribute__((ext_vector_type(8)))  float    v8f;
typedef __attribute__((ext_vector_type(2)))  _Float16 h2;
typedef __attribute__((ext_vector_type(4)))  int      v4i;

static __device__ __forceinline__ v8f wmma_f16(v16h a, v16h b, v8f c) {
  return __builtin_amdgcn_wmma_f32_16x16x32_f16(false, a, false, b, (short)0, c,
                                                false, false);
}

static __device__ __forceinline__ v16h ld_frag2(const _Float16* p0,
                                                const _Float16* p1) {
  union { v16h v; float4 q[2]; } u;
  u.q[0] = *(const float4*)p0;
  u.q[1] = *(const float4*)p1;
  return u.v;
}

// ---------------------------------------------------------------------------
// CDNA5 async global->LDS (guarded; falls back to register staging)
// Builtin signature on this toolchain wants v4i pointees:
//   (as(1) v4i* src, as(3) v4i* dst, imm offset, imm cpol)
// ---------------------------------------------------------------------------
#if defined(__has_builtin)
#  if __has_builtin(__builtin_amdgcn_global_load_async_to_lds_b128) && \
      __has_builtin(__builtin_amdgcn_s_wait_asynccnt)
#    define USE_ASYNC_LDS 1
#  endif
#endif
#ifndef USE_ASYNC_LDS
#  define USE_ASYNC_LDS 0
#endif

#if USE_ASYNC_LDS
typedef __attribute__((address_space(1))) v4i* gv4p;
typedef __attribute__((address_space(3))) v4i* lv4p;
static __device__ __forceinline__ gv4p to_glb(const void* p) {
  return (gv4p)(unsigned long long)p;
}
static __device__ __forceinline__ lv4p to_lds(void* p) {
  // generic 'shared' pointer keeps the LDS byte offset in its low 32 bits
  return (lv4p)(unsigned)(unsigned long long)p;
}
#endif

// ---------------------------------------------------------------------------
// Problem constants (fixed by the reference)
// ---------------------------------------------------------------------------
#define BSZ   2
#define LSEQ  2048
#define DIMM  1024
#define NHEAD 16
#define HDIM  64
#define DFFN  4096
#define TOK   (BSZ * LSEQ)   // 4096 rows

// ---------------------------------------------------------------------------
// f32 -> f16 cast (weights)
// ---------------------------------------------------------------------------
__global__ __launch_bounds__(256) void cast_f16_kernel(
    const float* __restrict__ s, _Float16* __restrict__ d, int n) {
  int i = (blockIdx.x * 256 + threadIdx.x) * 4;
  if (i + 3 < n) {
    float4 v = *(const float4*)(s + i);
    d[i + 0] = (_Float16)v.x;
    d[i + 1] = (_Float16)v.y;
    d[i + 2] = (_Float16)v.z;
    d[i + 3] = (_Float16)v.w;
  }
}

// ---------------------------------------------------------------------------
// RMSNorm: one 1024-col row per 256-thread block, f16 output
// ---------------------------------------------------------------------------
__global__ __launch_bounds__(256) void rmsnorm_kernel(
    const float* __restrict__ x, const float* __restrict__ w,
    _Float16* __restrict__ out, int ncols) {
  int row = blockIdx.x;
  const float* xr = x + (size_t)row * ncols;
  float ss = 0.f;
  for (int i = threadIdx.x; i < ncols; i += 256) { float v = xr[i]; ss += v * v; }
#pragma unroll
  for (int off = 16; off; off >>= 1) ss += __shfl_xor(ss, off);
  __shared__ float acc[8];
  if ((threadIdx.x & 31) == 0) acc[threadIdx.x >> 5] = ss;
  __syncthreads();
  float tot = 0.f;
#pragma unroll
  for (int i = 0; i < 8; ++i) tot += acc[i];
  float sc = rsqrtf(tot / (float)ncols + 1e-6f);
  for (int i = threadIdx.x; i < ncols; i += 256)
    out[(size_t)row * ncols + i] = (_Float16)(xr[i] * sc * w[i]);
}

// ---------------------------------------------------------------------------
// WMMA GEMM: C[MxN] = A[MxK] * B[KxN]  (A,B f16 row-major)
// Block tile 128x128x32, 256 threads = 8 waves (4x2), wave tile 32x64.
// Double-buffered LDS; A tile staged via async global->LDS when available,
// B tile register-prefetched and stored transposed. One barrier per K-step.
// ---------------------------------------------------------------------------
__global__ __launch_bounds__(256) void gemm_f16_kernel(
    const _Float16* __restrict__ A, const _Float16* __restrict__ B,
    float* __restrict__ Cf, _Float16* __restrict__ Ch,
    const float* __restrict__ Res, int M, int N, int K) {
  __shared__ _Float16 Al[2][128][40];   // [buf][m][k], padded rows (80B)
  __shared__ _Float16 Bt[2][128][40];   // [buf][n][k] (transposed), padded

  const int tid = threadIdx.x;
  const int lane = tid & 31, w = tid >> 5;
  const int wm = w >> 1, wn = w & 1;            // 4 x 2 wave grid
  const int l16 = lane & 15, ksel = lane >> 4;
  const int tM = blockIdx.y * 128, tN = blockIdx.x * 128;

  // A staging: this thread owns rows ra0 and ra0+64, 8-half chunk at ka0
  const int ra0 = tid >> 2, ka0 = (tid & 3) << 3;
  // B staging: this thread owns K-row pair (krp, krp+1), 8 cols at nc
  const int krp = (tid >> 4) << 1, nc = (tid & 15) << 3;

  float4 bR0, bR1;
#if !USE_ASYNC_LDS
  float4 aR0, aR1;
#endif

  auto load_B = [&](int kb) {
    bR0 = *(const float4*)(B + (size_t)(kb + krp) * N + tN + nc);
    bR1 = *(const float4*)(B + (size_t)(kb + krp + 1) * N + tN + nc);
  };
  auto store_B = [&](int buf) {
    const _Float16* p0 = (const _Float16*)&bR0;
    const _Float16* p1 = (const _Float16*)&bR1;
#pragma unroll
    for (int i = 0; i < 8; ++i) {
      h2 pr; pr.x = p0[i]; pr.y = p1[i];
      *(h2*)(&Bt[buf][nc + i][krp]) = pr;    // one ds_store_b32 per pair
    }
  };
  auto load_A = [&](int kb, int buf) {
#if USE_ASYNC_LDS
    __builtin_amdgcn_global_load_async_to_lds_b128(
        to_glb(A + (size_t)(tM + ra0) * K + kb + ka0),
        to_lds(&Al[buf][ra0][ka0]), 0, 0);
    __builtin_amdgcn_global_load_async_to_lds_b128(
        to_glb(A + (size_t)(tM + ra0 + 64) * K + kb + ka0),
        to_lds(&Al[buf][ra0 + 64][ka0]), 0, 0);
#else
    (void)buf;
    aR0 = *(const float4*)(A + (size_t)(tM + ra0) * K + kb + ka0);
    aR1 = *(const float4*)(A + (size_t)(tM + ra0 + 64) * K + kb + ka0);
#endif
  };
#if !USE_ASYNC_LDS
  auto store_A = [&](int buf) {
    *(float4*)(&Al[buf][ra0][ka0]) = aR0;
    *(float4*)(&Al[buf][ra0 + 64][ka0]) = aR1;
  };
#endif

  v8f acc[2][4];
#pragma unroll
  for (int m = 0; m < 2; ++m)
#pragma unroll
    for (int n = 0; n < 4; ++n) acc[m][n] = (v8f)0.f;

  // prologue: stage tile 0 into buffer 0
  load_A(0, 0);
  load_B(0);
  store_B(0);
#if USE_ASYNC_LDS
  __builtin_amdgcn_s_wait_asynccnt(0);
#else
  store_A(0);
#endif
  __syncthreads();

  int cur = 0;
  for (int kb = 0; kb < K; kb += 32) {
    const int nxt = cur ^ 1;
    const bool more = (kb + 32) < K;
    if (more) {           // prefetch next tile; overlaps with WMMAs below
      load_A(kb + 32, nxt);
      load_B(kb + 32);
    }

    v16h af[2], bf[4];
#pragma unroll
    for (int m = 0; m < 2; ++m) {
      const _Float16* p = &Al[cur][wm * 32 + m * 16 + l16][ksel * 8];
      af[m] = ld_frag2(p, p + 16);
    }
#pragma unroll
    for (int n = 0; n < 4; ++n) {
      const _Float16* p = &Bt[cur][wn * 64 + n * 16 + l16][ksel * 8];
      bf[n] = ld_frag2(p, p + 16);
    }
#pragma unroll
    for (int m = 0; m < 2; ++m)
#pragma unroll
      for (int n = 0; n < 4; ++n) acc[m][n] = wmma_f16(af[m], bf[n], acc[m][n]);

    if (more) {
      store_B(nxt);                       // waits the B global loads here
#if USE_ASYNC_LDS
      __builtin_amdgcn_s_wait_asynccnt(0);
#else
      store_A(nxt);
#endif
    }
    __syncthreads();
    cur = nxt;
  }

  // epilogue: C tile VGPR r holds row r + 8*ksel, col l16
  if (Ch) {
#pragma unroll
    for (int m = 0; m < 2; ++m)
#pragma unroll
      for (int n = 0; n < 4; ++n)
#pragma unroll
        for (int r = 0; r < 8; ++r) {
          size_t idx = (size_t)(tM + wm * 32 + m * 16 + r + (ksel << 3)) * N +
                       (tN + wn * 64 + n * 16 + l16);
          Ch[idx] = (_Float16)acc[m][n][r];
        }
  } else {
#pragma unroll
    for (int m = 0; m < 2; ++m)
#pragma unroll
      for (int n = 0; n < 4; ++n)
#pragma unroll
        for (int r = 0; r < 8; ++r) {
          size_t idx = (size_t)(tM + wm * 32 + m * 16 + r + (ksel << 3)) * N +
                       (tN + wn * 64 + n * 16 + l16);
          float v = acc[m][n][r];
          if (Res) v += Res[idx];
          Cf[idx] = v;
        }
  }
}

// ---------------------------------------------------------------------------
// Fused dual GEMM for FFN: out = (A*Wfc1) * gelu(A*Wgate), f16 output.
// Block tile 128x64x32, 256 threads = 8 waves (4x2), wave tile 32x32.
// Same double-buffered pipeline as gemm_f16_kernel.
// ---------------------------------------------------------------------------
__global__ __launch_bounds__(256) void gemm_dual_gelu_kernel(
    const _Float16* __restrict__ A, const _Float16* __restrict__ Bg,
    const _Float16* __restrict__ Bf, _Float16* __restrict__ Out,
    int M, int N, int K) {
  __shared__ _Float16 Al[2][128][40];
  __shared__ _Float16 Btg[2][64][40];
  __shared__ _Float16 Btf[2][64][40];

  const int tid = threadIdx.x;
  const int lane = tid & 31, w = tid >> 5;
  const int wm = w >> 1, wn = w & 1;
  const int l16 = lane & 15, ksel = lane >> 4;
  const int tM = blockIdx.y * 128, tN = blockIdx.x * 64;

  const int ra0 = tid >> 2, ka0 = (tid & 3) << 3;
  // B: 128 (pair,col-chunk) slots per matrix; threads 0-127 -> gate, 128-255 -> fc1
  const int krp = ((tid & 127) >> 3) << 1, nc = (tid & 7) << 3;
  const _Float16* Bsel = (tid >> 7) ? Bf : Bg;

  float4 bR0, bR1;
#if !USE_ASYNC_LDS
  float4 aR0, aR1;
#endif

  auto load_B = [&](int kb) {
    bR0 = *(const float4*)(Bsel + (size_t)(kb + krp) * N + tN + nc);
    bR1 = *(const float4*)(Bsel + (size_t)(kb + krp + 1) * N + tN + nc);
  };
  auto store_B = [&](int buf) {
    _Float16(*Bts)[40] = (tid >> 7) ? Btf[buf] : Btg[buf];
    const _Float16* p0 = (const _Float16*)&bR0;
    const _Float16* p1 = (const _Float16*)&bR1;
#pragma unroll
    for (int i = 0; i < 8; ++i) {
      h2 pr; pr.x = p0[i]; pr.y = p1[i];
      *(h2*)(&Bts[nc + i][krp]) = pr;
    }
  };
  auto load_A = [&](int kb, int buf) {
#if USE_ASYNC_LDS
    __builtin_amdgcn_global_load_async_to_lds_b128(
        to_glb(A + (size_t)(tM + ra0) * K + kb + ka0),
        to_lds(&Al[buf][ra0][ka0]), 0, 0);
    __builtin_amdgcn_global_load_async_to_lds_b128(
        to_glb(A + (size_t)(tM + ra0 + 64) * K + kb + ka0),
        to_lds(&Al[buf][ra0 + 64][ka0]), 0, 0);
#else
    (void)buf;
    aR0 = *(const float4*)(A + (size_t)(tM + ra0) * K + kb + ka0);
    aR1 = *(const float4*)(A + (size_t)(tM + ra0 + 64) * K + kb + ka0);
#endif
  };
#if !USE_ASYNC_LDS
  auto store_A = [&](int buf) {
    *(float4*)(&Al[buf][ra0][ka0]) = aR0;
    *(float4*)(&Al[buf][ra0 + 64][ka0]) = aR1;
  };
#endif

  v8f ag[2][2], af_[2][2];
#pragma unroll
  for (int m = 0; m < 2; ++m)
#pragma unroll
    for (int n = 0; n < 2; ++n) { ag[m][n] = (v8f)0.f; af_[m][n] = (v8f)0.f; }

  load_A(0, 0);
  load_B(0);
  store_B(0);
#if USE_ASYNC_LDS
  __builtin_amdgcn_s_wait_asynccnt(0);
#else
  store_A(0);
#endif
  __syncthreads();

  int cur = 0;
  for (int kb = 0; kb < K; kb += 32) {
    const int nxt = cur ^ 1;
    const bool more = (kb + 32) < K;
    if (more) { load_A(kb + 32, nxt); load_B(kb + 32); }

    v16h amf[2], bgf[2], bff[2];
#pragma unroll
    for (int m = 0; m < 2; ++m) {
      const _Float16* p = &Al[cur][wm * 32 + m * 16 + l16][ksel * 8];
      amf[m] = ld_frag2(p, p + 16);
    }
#pragma unroll
    for (int n = 0; n < 2; ++n) {
      const _Float16* pg = &Btg[cur][wn * 32 + n * 16 + l16][ksel * 8];
      const _Float16* pf = &Btf[cur][wn * 32 + n * 16 + l16][ksel * 8];
      bgf[n] = ld_frag2(pg, pg + 16);
      bff[n] = ld_frag2(pf, pf + 16);
    }
#pragma unroll
    for (int m = 0; m < 2; ++m)
#pragma unroll
      for (int n = 0; n < 2; ++n) {
        ag[m][n]  = wmma_f16(amf[m], bgf[n], ag[m][n]);
        af_[m][n] = wmma_f16(amf[m], bff[n], af_[m][n]);
      }

    if (more) {
      store_B(nxt);
#if USE_ASYNC_LDS
      __builtin_amdgcn_s_wait_asynccnt(0);
#else
      store_A(nxt);
#endif
    }
    __syncthreads();
    cur = nxt;
  }

#pragma unroll
  for (int m = 0; m < 2; ++m)
#pragma unroll
    for (int n = 0; n < 2; ++n)
#pragma unroll
      for (int r = 0; r < 8; ++r) {
        size_t idx = (size_t)(tM + wm * 32 + m * 16 + r + (ksel << 3)) * N +
                     (tN + wn * 32 + n * 16 + l16);
        float g = ag[m][n][r];
        float f = af_[m][n][r];
        float gl = 0.5f * g * (1.f + tanhf(0.7978845608f * (g + 0.044715f * g * g * g)));
        Out[idx] = (_Float16)(f * gl);
      }
}

// ---------------------------------------------------------------------------
// Flash attention with T5 relative-position bias.
// Grid (L/64, NH, B), 128 threads = 4 waves; wave owns 16 query rows.
// ---------------------------------------------------------------------------
static __device__ __forceinline__ int t5_bucket(int rel /* j - i */) {
  int bkt = (rel > 0) ? 16 : 0;
  int r = (rel < 0) ? -rel : rel;
  if (r < 8) return bkt + r;
  // 8 + log(r/8)/log(16)*8, clamped to 15
  int lg = 8 + (int)(__logf((float)r * 0.125f) * 2.8853900817779268f);
  return bkt + (lg < 15 ? lg : 15);
}

__global__ __launch_bounds__(128) void attn_kernel(
    const _Float16* __restrict__ Q, const _Float16* __restrict__ Kg,
    const _Float16* __restrict__ Vg, const int* __restrict__ mask,
    const float* __restrict__ rel_embed, _Float16* __restrict__ O) {
  __shared__ _Float16 Kl[32][72];       // [kv-row][c] row-major == transposed B
  __shared__ _Float16 Vt[64][40];       // [c][kv-row], transposed
  __shared__ _Float16 Pl[4][16][40];    // per-wave P staging (16x32)
  __shared__ float    btab[32];
  __shared__ int      mloc[32];

  const int tid = threadIdx.x, lane = tid & 31, w = tid >> 5;
  const int l16 = lane & 15, ksel = lane >> 4;
  const int b = blockIdx.z, h = blockIdx.y;
  const int i0 = blockIdx.x * 64 + w * 16;

  if (tid < 32) btab[tid] = rel_embed[tid * NHEAD + h];

  // resident Q fragments for this wave's 16 rows (two K-halves of c=64)
  v16h qf[2];
  {
    const _Float16* qrow = Q + ((size_t)(b * LSEQ + i0 + l16)) * DIMM + h * HDIM;
#pragma unroll
    for (int kh = 0; kh < 2; ++kh)
      qf[kh] = ld_frag2(qrow + kh * 32 + ksel * 8, qrow + kh * 32 + 16 + ksel * 8);
  }

  // tile prefetch registers: K chunks (rows jrA, jrA+16), V row-pair (jpV,jpV+1)
  const int jrA = tid >> 3, ccA = (tid & 7) << 3;
  const int jpV = (tid >> 3) << 1, ccV = (tid & 7) << 3;
  float4 kR0, kR1, vR0, vR1;
  int mR = 1;
  auto load_tile = [&](int j0) {
    const _Float16* kb_ = Kg + ((size_t)(b * LSEQ + j0)) * DIMM + h * HDIM;
    kR0 = *(const float4*)(kb_ + (size_t)jrA * DIMM + ccA);
    kR1 = *(const float4*)(kb_ + (size_t)(jrA + 16) * DIMM + ccA);
    const _Float16* vb_ = Vg + ((size_t)(b * LSEQ + j0)) * DIMM + h * HDIM;
    vR0 = *(const float4*)(vb_ + (size_t)jpV * DIMM + ccV);
    vR1 = *(const float4*)(vb_ + (size_t)(jpV + 1) * DIMM + ccV);
    if (tid < 32) mR = mask[b * LSEQ + j0 + tid];
  };
  auto store_tile = [&]() {
    *(float4*)(&Kl[jrA][ccA]) = kR0;
    *(float4*)(&Kl[jrA + 16][ccA]) = kR1;
    const _Float16* q0 = (const _Float16*)&vR0;
    const _Float16* q1 = (const _Float16*)&vR1;
#pragma unroll
    for (int i = 0; i < 8; ++i) {
      h2 pr; pr.x = q0[i]; pr.y = q1[i];
      *(h2*)(&Vt[ccV + i][jpV]) = pr;
    }
    if (tid < 32) mloc[tid] = mR;
  };

  float mrow[8], lrow[8];
  v8f o[4];
#pragma unroll
  for (int t = 0; t < 4; ++t) o[t] = (v8f)0.f;
#pragma unroll
  for (int r = 0; r < 8; ++r) { mrow[r] = -INFINITY; lrow[r] = 0.f; }

  load_tile(0);
  for (int j0 = 0; j0 < LSEQ; j0 += 32) {
    __syncthreads();              // previous iteration finished reading Kl/Vt
    store_tile();
    if (j0 + 32 < LSEQ) load_tile(j0 + 32);   // prefetch overlaps compute
    __syncthreads();

    // S = Q*K^T + bias (bias & mask folded into WMMA C-init)
    v8f s[2];
#pragma unroll
    for (int jj = 0; jj < 2; ++jj) {
#pragma unroll
      for (int r = 0; r < 8; ++r) {
        int j = j0 + jj * 16 + l16;
        int i = i0 + r + (ksel << 3);
        float bv = btab[t5_bucket(j - i)];
        s[jj][r] = (mloc[jj * 16 + l16] == 0) ? -3.402823466e38f : bv;
      }
#pragma unroll
      for (int kh = 0; kh < 2; ++kh) {
        const _Float16* p = &Kl[jj * 16 + l16][kh * 32 + ksel * 8];
        v16h bf = ld_frag2(p, p + 16);
        s[jj] = wmma_f16(qf[kh], bf, s[jj]);
      }
    }

    // online softmax over the 32 new columns (16-lane row groups, wave32)
#pragma unroll
    for (int r = 0; r < 8; ++r) {
      float v0 = s[0][r], v1 = s[1][r];
      float mx = fmaxf(v0, v1);
#pragma unroll
      for (int off = 1; off < 16; off <<= 1) mx = fmaxf(mx, __shfl_xor(mx, off));
      float nm = fmaxf(mrow[r], mx);
      float sc = __expf(mrow[r] - nm);
      float p0 = __expf(v0 - nm), p1 = __expf(v1 - nm);
      float ps = p0 + p1;
#pragma unroll
      for (int off = 1; off < 16; off <<= 1) ps += __shfl_xor(ps, off);
      lrow[r] = lrow[r] * sc + ps;
      mrow[r] = nm;
#pragma unroll
      for (int t = 0; t < 4; ++t) o[t][r] *= sc;   // rescale only row r
      int m = r + (ksel << 3);
      Pl[w][m][l16]      = (_Float16)p0;
      Pl[w][m][16 + l16] = (_Float16)p1;
    }
    // per-wave P staging: same-wave LDS RAW is in-order; block reordering only
    __builtin_amdgcn_wave_barrier();

    // O += P * V
    v16h pf;
    {
      const _Float16* pr = &Pl[w][l16][0];
      pf = ld_frag2(pr + ksel * 8, pr + 16 + ksel * 8);
    }
#pragma unroll
    for (int t = 0; t < 4; ++t) {
      const _Float16* vp = &Vt[t * 16 + l16][ksel * 8];
      v16h vf = ld_frag2(vp, vp + 16);
      o[t] = wmma_f16(pf, vf, o[t]);
    }
  }

  // normalize and store (f16, feeds the Wo projection GEMM)
#pragma unroll
  for (int t = 0; t < 4; ++t)
#pragma unroll
    for (int r = 0; r < 8; ++r) {
      int row = i0 + r + (ksel << 3);
      O[((size_t)(b * LSEQ + row)) * DIMM + h * HDIM + t * 16 + l16] =
          (_Float16)(o[t][r] / lrow[r]);
    }
}

// ---------------------------------------------------------------------------
// Launch
// ---------------------------------------------------------------------------
extern "C" void kernel_launch(void* const* d_in, const int* in_sizes, int n_in,
                              void* d_out, int out_size, void* d_ws, size_t ws_size,
                              hipStream_t stream) {
  const float* x    = (const float*)d_in[0];
  const int*   mask = (const int*)d_in[1];
  const float* n1w  = (const float*)d_in[2];
  const float* wq   = (const float*)d_in[3];
  const float* wk   = (const float*)d_in[4];
  const float* wv   = (const float*)d_in[5];
  const float* wo   = (const float*)d_in[6];
  const float* n2w  = (const float*)d_in[7];
  const float* gw   = (const float*)d_in[8];
  const float* f1w  = (const float*)d_in[9];
  const float* f2w  = (const float*)d_in[10];
  const float* rel  = (const float*)d_in[11];

  // workspace carve-up (~128 MB, all write-before-read)
  size_t off = 0;
  char* ws = (char*)d_ws;
  auto nxt = [&](size_t bytes) -> void* {
    void* p = ws + off;
    off += (bytes + 255) & ~(size_t)255;
    return p;
  };
  _Float16* hw    = (_Float16*)nxt((size_t)TOK * DIMM * 2);
  _Float16* wq16  = (_Float16*)nxt((size_t)DIMM * DIMM * 2);
  _Float16* wk16  = (_Float16*)nxt((size_t)DIMM * DIMM * 2);
  _Float16* wv16  = (_Float16*)nxt((size_t)DIMM * DIMM * 2);
  _Float16* wo16  = (_Float16*)nxt((size_t)DIMM * DIMM * 2);
  _Float16* gw16  = (_Float16*)nxt((size_t)DIMM * DFFN * 2);
  _Float16* f1w16 = (_Float16*)nxt((size_t)DIMM * DFFN * 2);
  _Float16* f2w16 = (_Float16*)nxt((size_t)DFFN * DIMM * 2);
  _Float16* q16   = (_Float16*)nxt((size_t)TOK * DIMM * 2);
  _Float16* k16   = (_Float16*)nxt((size_t)TOK * DIMM * 2);
  _Float16* v16   = (_Float16*)nxt((size_t)TOK * DIMM * 2);
  _Float16* o16   = (_Float16*)nxt((size_t)TOK * DIMM * 2);
  float*    x1    = (float*)   nxt((size_t)TOK * DIMM * 4);
  _Float16* h2v   = (_Float16*)nxt((size_t)TOK * DIMM * 2);
  _Float16* ff16  = (_Float16*)nxt((size_t)TOK * DFFN * 2);

  auto cast = [&](const float* s, _Float16* d, size_t n) {
    int blocks = (int)((n / 4 + 255) / 256);
    cast_f16_kernel<<<blocks, 256, 0, stream>>>(s, d, (int)n);
  };
  cast(wq, wq16, (size_t)DIMM * DIMM);
  cast(wk, wk16, (size_t)DIMM * DIMM);
  cast(wv, wv16, (size_t)DIMM * DIMM);
  cast(wo, wo16, (size_t)DIMM * DIMM);
  cast(gw, gw16, (size_t)DIMM * DFFN);
  cast(f1w, f1w16, (size_t)DIMM * DFFN);
  cast(f2w, f2w16, (size_t)DFFN * DIMM);

  // h = rmsnorm(x, norm1_w)
  rmsnorm_kernel<<<TOK, 256, 0, stream>>>(x, n1w, hw, DIMM);

  // Q, K, V projections (f16 out)
  {
    dim3 grid(DIMM / 128, TOK / 128);
    gemm_f16_kernel<<<grid, 256, 0, stream>>>(hw, wq16, nullptr, q16, nullptr,
                                              TOK, DIMM, DIMM);
    gemm_f16_kernel<<<grid, 256, 0, stream>>>(hw, wk16, nullptr, k16, nullptr,
                                              TOK, DIMM, DIMM);
    gemm_f16_kernel<<<grid, 256, 0, stream>>>(hw, wv16, nullptr, v16, nullptr,
                                              TOK, DIMM, DIMM);
  }

  // attention
  {
    dim3 grid(LSEQ / 64, NHEAD, BSZ);
    attn_kernel<<<grid, 128, 0, stream>>>(q16, k16, v16, mask, rel, o16);
  }

  // x1 = x + O * Wo   (f32 out + residual)
  {
    dim3 grid(DIMM / 128, TOK / 128);
    gemm_f16_kernel<<<grid, 256, 0, stream>>>(o16, wo16, x1, nullptr, x,
                                              TOK, DIMM, DIMM);
  }

  // h2 = rmsnorm(x1, norm2_w)
  rmsnorm_kernel<<<TOK, 256, 0, stream>>>(x1, n2w, h2v, DIMM);

  // ff = fc1(h2) * gelu(gate(h2))  (f16 out)
  {
    dim3 grid(DFFN / 64, TOK / 128);
    gemm_dual_gelu_kernel<<<grid, 256, 0, stream>>>(h2v, gw16, f1w16, ff16,
                                                    TOK, DFFN, DIMM);
  }

  // out = x1 + ff * Wfc2  (f32 out + residual) -> d_out
  {
    dim3 grid(DIMM / 128, TOK / 128);
    gemm_f16_kernel<<<grid, 256, 0, stream>>>(ff16, f2w16, (float*)d_out,
                                              nullptr, x1, TOK, DIMM, DFFN);
  }
}
